// ImprovedGATModel_53824530153869
// MI455X (gfx1250) — matrix operand order repaired and checked by reading, not compiled, not run on hardware
//
#include <hip/hip_runtime.h>
#include <math.h>

#define N_NODES   50000
#define N_EDGES   800000
#define HID       128
#define N_GRAPHS  32
#define EDGES_TOT (N_EDGES + N_NODES)   // self loops appended
#define NEG_SLOPE 0.2f
#define LN_EPS    1e-5f

typedef __attribute__((ext_vector_type(2)))  float  v2f;
typedef __attribute__((ext_vector_type(8)))  float  v8f;
typedef __attribute__((ext_vector_type(16))) __bf16 v16bf;

// ---------------------------------------------------------------------------
// float atomic max via sign-split integer atomics (monotone under races)
// ---------------------------------------------------------------------------
__device__ __forceinline__ void atomicMaxF32(float* addr, float val) {
  if (val >= 0.0f) atomicMax((int*)addr, __float_as_int(val));
  else             atomicMin((unsigned int*)addr, __float_as_uint(val));
}

// ---------------------------------------------------------------------------
// C[M x 128] = A[M x 128] * B[128 x 128] (+bias), fp32 WMMA 16x16x4
// grid.x = M/16 tiles, block = 256 (8 waves; wave w owns N-tile w)
// ---------------------------------------------------------------------------
__global__ void gemm128_wmma(const float* __restrict__ A,
                             const float* __restrict__ B,
                             const float* __restrict__ bias,
                             float* __restrict__ C) {
  const int lane = threadIdx.x & 31;
  const int wave = threadIdx.x >> 5;        // 0..7 -> N tile
  const int row0 = blockIdx.x * 16;
  const int mr   = lane & 15;
  const int hi   = lane >> 4;               // lane group (K half)
  const int ncol = wave * 16 + mr;          // B/C column for this lane
  const float* Arow = A + (size_t)(row0 + mr) * HID;

  v8f acc = {0.f, 0.f, 0.f, 0.f, 0.f, 0.f, 0.f, 0.f};

#if __has_builtin(__builtin_amdgcn_wmma_f32_16x16x4_f32)
  // fp32 path: A 16x4 frag: lane holds (M=mr, K=kb+2*hi, kb+2*hi+1)
  //            B  4x16 frag: lane holds (K=kb+2*hi{,+1}, N=ncol)
  for (int kb = 0; kb < HID; kb += 4) {
    const int k = kb + 2 * hi;
    v2f a; a.x = Arow[k];                  a.y = Arow[k + 1];
    v2f b; b.x = B[(size_t)k * HID + ncol]; b.y = B[(size_t)(k + 1) * HID + ncol];
    acc = __builtin_amdgcn_wmma_f32_16x16x4_f32(false, a, false, b,
                                                (short)0, acc, false, false);
  }
#else
  // fallback: bf16 16x16x32 with f32 accumulate (codegen-confirmed builtin)
  for (int kb = 0; kb < HID; kb += 32) {
    v16bf a, b;
#pragma unroll
    for (int v = 0; v < 4; ++v) {
      int k0 = kb + hi * 8 + 2 * v;        // K=0..15 half
      a[2 * v]     = (__bf16)Arow[k0];
      a[2 * v + 1] = (__bf16)Arow[k0 + 1];
      int k1 = kb + 16 + hi * 8 + 2 * v;   // K=16..31 half
      a[8 + 2 * v]     = (__bf16)Arow[k1];
      a[8 + 2 * v + 1] = (__bf16)Arow[k1 + 1];
    }
#pragma unroll
    for (int s = 0; s < 16; ++s) {
      int k = kb + hi * 8 + (s & 7) + (s >> 3) * 16;
      b[s] = (__bf16)B[(size_t)k * HID + ncol];
    }
    acc = __builtin_amdgcn_wmma_f32_16x16x32_bf16(false, a, false, b,
                                                  (short)0, acc, false, false);
  }
#endif

  const float bv = bias ? bias[ncol] : 0.0f;
  const int mb = hi * 8;                    // C/D: lanes 16-31 hold M=8..15
#pragma unroll
  for (int v = 0; v < 8; ++v)
    C[(size_t)(row0 + mb + v) * HID + ncol] = acc[v] + bv;
}

// ---------------------------------------------------------------------------
// per-node attention dots: e_src[n] = h[n].a_s ; e_dst[n] = h[n].a_d
// one wave per node
// ---------------------------------------------------------------------------
__global__ void node_attn_dots(const float* __restrict__ h,
                               const float* __restrict__ a_s,
                               const float* __restrict__ a_d,
                               float* __restrict__ es,
                               float* __restrict__ ed) {
  const int wave = (blockIdx.x * blockDim.x + threadIdx.x) >> 5;
  const int lane = threadIdx.x & 31;
  if (wave >= N_NODES) return;
  const float4 hv  = *(const float4*)(h   + (size_t)wave * HID + lane * 4);
  const float4 s4  = *(const float4*)(a_s + lane * 4);
  const float4 d4  = *(const float4*)(a_d + lane * 4);
  float ss = hv.x * s4.x + hv.y * s4.y + hv.z * s4.z + hv.w * s4.w;
  float sd = hv.x * d4.x + hv.y * d4.y + hv.z * d4.z + hv.w * d4.w;
#pragma unroll
  for (int o = 1; o < 32; o <<= 1) {
    ss += __shfl_xor(ss, o, 32);
    sd += __shfl_xor(sd, o, 32);
  }
  if (lane == 0) { es[wave] = ss; ed[wave] = sd; }
}

__global__ void init_mz(float* __restrict__ m, float* __restrict__ z) {
  int i = blockIdx.x * blockDim.x + threadIdx.x;
  if (i < N_NODES) { m[i] = -3.402823466e38f; z[i] = 0.0f; }
}

// per-edge: leaky-relu logit + segment max over dst
__global__ void edge_logits(const int* __restrict__ src, const int* __restrict__ dst,
                            const float* __restrict__ es, const float* __restrict__ ed,
                            float* __restrict__ ebuf, float* __restrict__ m) {
  int e = blockIdx.x * blockDim.x + threadIdx.x;
  if (e >= EDGES_TOT) return;
  int s, d;
  if (e < N_EDGES) { s = src[e]; d = dst[e]; } else { s = d = e - N_EDGES; }
  float v = es[s] + ed[d];
  v = v > 0.0f ? v : NEG_SLOPE * v;
  ebuf[e] = v;
  atomicMaxF32(m + d, v);
}

// per-edge: p = exp(e - m[dst]); segment sum
__global__ void edge_exp(const int* __restrict__ dst,
                         const float* __restrict__ ebuf, const float* __restrict__ m,
                         float* __restrict__ pbuf, float* __restrict__ z) {
  int e = blockIdx.x * blockDim.x + threadIdx.x;
  if (e >= EDGES_TOT) return;
  int d = (e < N_EDGES) ? dst[e] : (e - N_EDGES);
  float p = expf(ebuf[e] - m[d]);
  pbuf[e] = p;
  atomicAdd(z + d, p);
}

// per-edge weighted scatter: agg[dst] += alpha * h[src]  (one wave per edge)
__global__ void edge_scatter(const int* __restrict__ src, const int* __restrict__ dst,
                             const float* __restrict__ h,
                             const float* __restrict__ pbuf, const float* __restrict__ z,
                             float* __restrict__ agg) {
  const int wave = (blockIdx.x * blockDim.x + threadIdx.x) >> 5;
  const int lane = threadIdx.x & 31;
  if (wave >= EDGES_TOT) return;
  int s, d;
  if (wave < N_EDGES) { s = src[wave]; d = dst[wave]; } else { s = d = wave - N_EDGES; }
  const float alpha = pbuf[wave] / z[d];
  const float4 hv = *(const float4*)(h + (size_t)s * HID + lane * 4);
  float* out = agg + (size_t)d * HID + lane * 4;
  atomicAdd(out + 0, hv.x * alpha);
  atomicAdd(out + 1, hv.y * alpha);
  atomicAdd(out + 2, hv.z * alpha);
  atomicAdd(out + 3, hv.w * alpha);
}

// per-node: +bias, layernorm, optional elu, + residual  (one wave per node)
__global__ void ln_residual(const float* __restrict__ agg, const float* __restrict__ gb,
                            const float* __restrict__ lg, const float* __restrict__ lb,
                            const float* __restrict__ resid, float* __restrict__ out,
                            int do_elu) {
  const int wave = (blockIdx.x * blockDim.x + threadIdx.x) >> 5;
  const int lane = threadIdx.x & 31;
  if (wave >= N_NODES) return;
  float4 v  = *(const float4*)(agg + (size_t)wave * HID + lane * 4);
  float4 b0 = *(const float4*)(gb + lane * 4);
  v.x += b0.x; v.y += b0.y; v.z += b0.z; v.w += b0.w;
  float s  = v.x + v.y + v.z + v.w;
  float sq = v.x * v.x + v.y * v.y + v.z * v.z + v.w * v.w;
#pragma unroll
  for (int o = 1; o < 32; o <<= 1) {
    s  += __shfl_xor(s, o, 32);
    sq += __shfl_xor(sq, o, 32);
  }
  const float mean = s * (1.0f / HID);
  const float var  = sq * (1.0f / HID) - mean * mean;
  const float rs   = rsqrtf(var + LN_EPS);
  const float4 g4 = *(const float4*)(lg + lane * 4);
  const float4 b4 = *(const float4*)(lb + lane * 4);
  const float4 r4 = *(const float4*)(resid + (size_t)wave * HID + lane * 4);
  float y[4] = {(v.x - mean) * rs * g4.x + b4.x,
                (v.y - mean) * rs * g4.y + b4.y,
                (v.z - mean) * rs * g4.z + b4.z,
                (v.w - mean) * rs * g4.w + b4.w};
  if (do_elu) {
#pragma unroll
    for (int j = 0; j < 4; ++j) y[j] = y[j] > 0.0f ? y[j] : (expf(y[j]) - 1.0f);
  }
  float4 o4 = {y[0] + r4.x, y[1] + r4.y, y[2] + r4.z, y[3] + r4.w};
  *(float4*)(out + (size_t)wave * HID + lane * 4) = o4;
}

__global__ void init_pool(float* __restrict__ maxa, float* __restrict__ cnt) {
  int i = blockIdx.x * blockDim.x + threadIdx.x;
  if (i < N_GRAPHS * HID) maxa[i] = -3.402823466e38f;
  if (i < N_GRAPHS)       cnt[i]  = 0.0f;
}

// graph pooling accumulators (one wave per node)
__global__ void pool_accum(const float* __restrict__ h, const int* __restrict__ batch,
                           float* __restrict__ meana, float* __restrict__ maxa,
                           float* __restrict__ cnt) {
  const int wave = (blockIdx.x * blockDim.x + threadIdx.x) >> 5;
  const int lane = threadIdx.x & 31;
  if (wave >= N_NODES) return;
  const int g = batch[wave];
  const float4 hv = *(const float4*)(h + (size_t)wave * HID + lane * 4);
  float* ma = meana + (size_t)g * HID + lane * 4;
  float* mx = maxa  + (size_t)g * HID + lane * 4;
  atomicAdd(ma + 0, hv.x); atomicMaxF32(mx + 0, hv.x);
  atomicAdd(ma + 1, hv.y); atomicMaxF32(mx + 1, hv.y);
  atomicAdd(ma + 2, hv.z); atomicMaxF32(mx + 2, hv.z);
  atomicAdd(ma + 3, hv.w); atomicMaxF32(mx + 3, hv.w);
  if (lane == 0) atomicAdd(cnt + g, 1.0f);
}

// tiny readout MLP: [32,256] -> relu 128 -> relu 64 -> 4   (one block)
__global__ void readout_mlp(const float* __restrict__ meana, const float* __restrict__ maxa,
                            const float* __restrict__ cnt,
                            const float* __restrict__ W1, const float* __restrict__ b1,
                            const float* __restrict__ W2, const float* __restrict__ b2,
                            const float* __restrict__ W3, const float* __restrict__ b3,
                            float* __restrict__ out) {
  __shared__ float g[N_GRAPHS * 256];
  __shared__ float h1[N_GRAPHS * 128];
  __shared__ float h2[N_GRAPHS * 64];
  const int tid = threadIdx.x;
  for (int i = tid; i < N_GRAPHS * 256; i += 256) {
    int gi = i >> 8, c = i & 255;
    g[i] = (c < HID) ? meana[gi * HID + c] / fmaxf(cnt[gi], 1.0f)
                     : maxa[gi * HID + (c - HID)];
  }
  __syncthreads();
  for (int i = tid; i < N_GRAPHS * 128; i += 256) {
    int gi = i >> 7, n = i & 127;
    float acc = b1[n];
    for (int k = 0; k < 256; ++k) acc += g[gi * 256 + k] * W1[k * 128 + n];
    h1[i] = fmaxf(acc, 0.0f);
  }
  __syncthreads();
  for (int i = tid; i < N_GRAPHS * 64; i += 256) {
    int gi = i >> 6, n = i & 63;
    float acc = b2[n];
    for (int k = 0; k < 128; ++k) acc += h1[gi * 128 + k] * W2[k * 64 + n];
    h2[i] = fmaxf(acc, 0.0f);
  }
  __syncthreads();
  for (int i = tid; i < N_GRAPHS * 4; i += 256) {
    int gi = i >> 2, n = i & 3;
    float acc = b3[n];
    for (int k = 0; k < 64; ++k) acc += h2[gi * 64 + k] * W3[k * 4 + n];
    out[i] = acc;
  }
}

// ---------------------------------------------------------------------------
extern "C" void kernel_launch(void* const* d_in, const int* in_sizes, int n_in,
                              void* d_out, int out_size, void* d_ws, size_t ws_size,
                              hipStream_t stream) {
  const float* x      = (const float*)d_in[0];
  const int*   ei     = (const int*)d_in[1];
  const int*   batch  = (const int*)d_in[2];
  const float* Ws     = (const float*)d_in[3];
  const float* a_src  = (const float*)d_in[4];
  const float* a_dst  = (const float*)d_in[5];
  const float* gat_b  = (const float*)d_in[6];
  const float* ln_g   = (const float*)d_in[7];
  const float* ln_b   = (const float*)d_in[8];
  const float* skip_W = (const float*)d_in[9];
  const float* skip_b = (const float*)d_in[10];
  const float* W1     = (const float*)d_in[11];
  const float* b1     = (const float*)d_in[12];
  const float* W2     = (const float*)d_in[13];
  const float* b2     = (const float*)d_in[14];
  const float* W3     = (const float*)d_in[15];
  const float* b3     = (const float*)d_in[16];
  const int* src = ei;            // edge_index[0]
  const int* dst = ei + N_EDGES;  // edge_index[1]

  const size_t NF = (size_t)N_NODES * HID;
  float* p = (float*)d_ws;
  float* skip = p;  p += NF;
  float* hbuf = p;  p += NF;
  float* agg  = p;  p += NF;
  float* stA  = p;  p += NF;
  float* stB  = p;  p += NF;
  float* es   = p;  p += N_NODES;
  float* edd  = p;  p += N_NODES;
  float* mseg = p;  p += N_NODES;
  float* zseg = p;  p += N_NODES;
  float* ebuf = p;  p += EDGES_TOT;
  float* pbuf = p;  p += EDGES_TOT;
  float* meana = p; p += (size_t)N_GRAPHS * HID;
  float* maxa  = p; p += (size_t)N_GRAPHS * HID;
  float* cnt   = p; p += N_GRAPHS;

  const dim3 blk(256);
  const int gemm_grid  = N_NODES / 16;                 // 3125 (exact)
  const int nodew_grid = N_NODES / 8;                  // 6250 waves-per-node grids
  const int edge_grid  = (EDGES_TOT + 255) / 256;      // scalar edge kernels
  const int edgew_grid = (EDGES_TOT + 7) / 8;          // wave-per-edge grid
  const int nmz_grid   = (N_NODES + 255) / 256;

  // skip projection
  gemm128_wmma<<<gemm_grid, blk, 0, stream>>>(x, skip_W, skip_b, skip);

  const float* lin[4]   = { x, stA, stB, stA };
  float*       lout[4]  = { stA, stB, stA, stB };

  for (int i = 0; i < 4; ++i) {
    const float* resid = (i == 0) ? skip : lin[i];
    gemm128_wmma<<<gemm_grid, blk, 0, stream>>>(lin[i], Ws + (size_t)i * HID * HID,
                                                nullptr, hbuf);
    node_attn_dots<<<nodew_grid, blk, 0, stream>>>(hbuf, a_src + i * HID,
                                                   a_dst + i * HID, es, edd);
    init_mz<<<nmz_grid, blk, 0, stream>>>(mseg, zseg);
    hipMemsetAsync(agg, 0, NF * sizeof(float), stream);
    edge_logits<<<edge_grid, blk, 0, stream>>>(src, dst, es, edd, ebuf, mseg);
    edge_exp<<<edge_grid, blk, 0, stream>>>(dst, ebuf, mseg, pbuf, zseg);
    edge_scatter<<<edgew_grid, blk, 0, stream>>>(src, dst, hbuf, pbuf, zseg, agg);
    ln_residual<<<nodew_grid, blk, 0, stream>>>(agg, gat_b + i * HID, ln_g + i * HID,
                                                ln_b + i * HID, resid, lout[i],
                                                (i < 3) ? 1 : 0);
  }

  hipMemsetAsync(meana, 0, (size_t)N_GRAPHS * HID * sizeof(float), stream);
  init_pool<<<(N_GRAPHS * HID + 255) / 256, blk, 0, stream>>>(maxa, cnt);
  pool_accum<<<nodew_grid, blk, 0, stream>>>(lout[3], batch, meana, maxa, cnt);
  readout_mlp<<<1, blk, 0, stream>>>(meana, maxa, cnt, W1, b1, W2, b2, W3, b3,
                                     (float*)d_out);
}
